// DoRA_LoKr_55155970015584
// MI455X (gfx1250) — matrix operand order repaired and checked by reading, not compiled, not run on hardware
//
#include <hip/hip_runtime.h>
#include <hip/hip_bf16.h>
#include <stdint.h>

// ---------------------------------------------------------------------------
// DoRA + LoKr fused as:  out = x @ Wt^T + cvec,  with
//   Wt[o,c] = (m[o]-1)*base_w[o,c] + m[o]*s * w1[p,i]*w2[q,j]
//   cvec[o] = (m[o]-1)*bias[o],  m[o] = dora[o]/||base_w[o,:] + s*adapter[:,o]||
// Main GEMM: bf16 WMMA (v_wmma_f32_16x16x32_bf16), double-buffered LDS,
// B tiles via global_load_async_to_lds_b128 (ASYNCcnt), one barrier/iter.
// ---------------------------------------------------------------------------

#define DIN   4096
#define DOUT  4096
#define MTOT  16384          // B*S = 4*4096
#define SCALE 0.25f          // alpha/r = 1/4

typedef __attribute__((ext_vector_type(16))) __bf16        v16bf;
typedef __attribute__((ext_vector_type(16))) unsigned short v16u;
typedef __attribute__((ext_vector_type(8)))  unsigned short v8u;
typedef __attribute__((ext_vector_type(8)))  float          v8f;

// Exact round-to-nearest-even (used once, in k_wt / k_norm path).
__device__ __forceinline__ unsigned short f2bf(float f) {
    union { float f; unsigned int u; } c; c.f = f;
    unsigned int u = c.u;
    u += 0x7fffu + ((u >> 16) & 1u);
    return (unsigned short)(u >> 16);
}

// Fast packed conversion for the GEMM hot loop: 2 adds + 1 v_perm_b32.
// Result = {bf16(y), bf16(x)} (x in low 16 bits), round-half-up.
__device__ __forceinline__ unsigned pack2(float x, float y) {
    unsigned a = __builtin_bit_cast(unsigned, x) + 0x8000u;
    unsigned b = __builtin_bit_cast(unsigned, y) + 0x8000u;
    return __builtin_amdgcn_perm(b, a, 0x07060302u);
}

// ---- Kernel 1: w1 = w1a@w1b (64x64), w2 = w2a@w2b (64x64) -----------------
__global__ void k_factors(const float* __restrict__ w1a, const float* __restrict__ w1b,
                          const float* __restrict__ w2a, const float* __restrict__ w2b,
                          float* __restrict__ w1, float* __restrict__ w2) {
    for (int idx = threadIdx.x; idx < 64 * 64; idx += blockDim.x) {
        int r = idx >> 6, c = idx & 63;
        float s1 = 0.f, s2 = 0.f;
#pragma unroll
        for (int k = 0; k < 4; ++k) {
            s1 += w1a[r * 4 + k] * w1b[k * 64 + c];
            s2 += w2a[r * 4 + k] * w2b[k * 64 + c];
        }
        w1[idx] = s1; w2[idx] = s2;
    }
}

// ---- Kernel 2: row norms -> mag scale m[o], cvec[o] -----------------------
__global__ __launch_bounds__(256)
void k_norm(const float* __restrict__ bw, const float* __restrict__ bias,
            const float* __restrict__ dora, const float* __restrict__ w1,
            const float* __restrict__ w2, float* __restrict__ mag,
            float* __restrict__ cvec) {
    __shared__ float red[256];
    const int o = blockIdx.x;
    const int p = o >> 6, q = o & 63;
    const int t = threadIdx.x;
    const int cbase = t * 16;                 // 16 contiguous columns per thread
    const int i = cbase >> 6;                 // constant within the 16-chunk
    const float w1pi = SCALE * w1[p * 64 + i];
    const float4* r4 = (const float4*)(bw + (size_t)o * DIN + cbase);
    float acc = 0.f;
#pragma unroll
    for (int v = 0; v < 4; ++v) {
        float4 b = r4[v];
        int j0 = (cbase + v * 4) & 63;
        float e0 = b.x + w1pi * w2[q * 64 + j0 + 0];
        float e1 = b.y + w1pi * w2[q * 64 + j0 + 1];
        float e2 = b.z + w1pi * w2[q * 64 + j0 + 2];
        float e3 = b.w + w1pi * w2[q * 64 + j0 + 3];
        acc += e0 * e0 + e1 * e1 + e2 * e2 + e3 * e3;
    }
    red[t] = acc;
    __syncthreads();
    for (int off = 128; off > 0; off >>= 1) {
        if (t < off) red[t] += red[t + off];
        __syncthreads();
    }
    if (t == 0) {
        float m = dora[o] / sqrtf(red[0]);
        mag[o]  = m;
        cvec[o] = (m - 1.f) * bias[o];
    }
}

// ---- Kernel 3: materialize effective weight Wt[o][c] in bf16 (exact RNE) --
__global__ __launch_bounds__(256)
void k_wt(const float* __restrict__ bw, const float* __restrict__ w1,
          const float* __restrict__ w2, const float* __restrict__ mag,
          unsigned short* __restrict__ wt) {
    const int o = blockIdx.x;
    const int p = o >> 6, q = o & 63;
    const float m   = mag[o];
    const float mm1 = m - 1.f;
    const int t = threadIdx.x;
    const int cbase = t * 16;
    const int i = cbase >> 6;
    const float a1 = m * SCALE * w1[p * 64 + i];
    const float4* r4 = (const float4*)(bw + (size_t)o * DIN + cbase);
    unsigned int pk[8];
#pragma unroll
    for (int v = 0; v < 4; ++v) {
        float4 b = r4[v];
        int j0 = (cbase + v * 4) & 63;
        float e0 = mm1 * b.x + a1 * w2[q * 64 + j0 + 0];
        float e1 = mm1 * b.y + a1 * w2[q * 64 + j0 + 1];
        float e2 = mm1 * b.z + a1 * w2[q * 64 + j0 + 2];
        float e3 = mm1 * b.w + a1 * w2[q * 64 + j0 + 3];
        pk[v * 2 + 0] = (unsigned)f2bf(e0) | ((unsigned)f2bf(e1) << 16);
        pk[v * 2 + 1] = (unsigned)f2bf(e2) | ((unsigned)f2bf(e3) << 16);
    }
    uint4* dst = (uint4*)(wt + (size_t)o * DIN + cbase);
    dst[0] = make_uint4(pk[0], pk[1], pk[2], pk[3]);
    dst[1] = make_uint4(pk[4], pk[5], pk[6], pk[7]);
}

// ---- Kernel 4: GEMM  out[M=16384,N=4096] = X(bf16) * Wt^T + cvec ----------
#define BM 128
#define BN 128
#define BK 32
#define LDT 40                 // padded LDS row stride (elements, 80B, 16B-aligned)

__global__ __launch_bounds__(256)
void k_gemm(const float* __restrict__ X, const unsigned short* __restrict__ Wt,
            const float* __restrict__ cvec, float* __restrict__ Out) {
    __shared__ unsigned short As[2][BM * LDT];   // [m][k], bf16 bits (double buffer)
    __shared__ unsigned short Bs[2][BN * LDT];   // [n][k], bf16 bits, k-contiguous

    const int nBase = blockIdx.x * BN;
    const int mBase = blockIdx.y * BM;
    const int t     = threadIdx.x;
    const int lane  = t & 31;
    const int wave  = t >> 5;                 // 8 waves: 4 (M) x 2 (N)
    const int wm    = (wave & 3) * 32;
    const int wn    = (wave >> 2) * 64;
    const int lrow  = t >> 1;                 // 0..127: row of the staged tile
    const int lhalf = (t & 1) * 16;           // 16-element k-chunk

    v8f acc[2][4];
#pragma unroll
    for (int a = 0; a < 2; ++a)
#pragma unroll
        for (int b = 0; b < 4; ++b) acc[a][b] = {};

    // --- B tile: async global->LDS copy (32 bytes/thread, 2x b128) --------
    auto asyncB = [&](int kt, int buf) {
        uint64_t ga = (uint64_t)(uintptr_t)(Wt + (size_t)(nBase + lrow) * DIN + kt * BK + lhalf);
        unsigned  la = (unsigned)(uintptr_t)&Bs[buf][lrow * LDT + lhalf];
        // INST_OFFSET applies to both LDS and global addresses (ISA 10.x).
        asm volatile("global_load_async_to_lds_b128 %0, %1, off\n\t"
                     "global_load_async_to_lds_b128 %0, %1, off offset:16"
                     :: "v"(la), "v"(ga) : "memory");
    };
    auto waitAsync = [&]() {
        asm volatile("s_wait_asynccnt 0x0" ::: "memory");
    };

    // --- A tile: global f32 -> regs ----------------------------------------
    auto loadA = [&](int kt, float4 ra[4]) {
        const float4* p = (const float4*)(X + (size_t)(mBase + lrow) * DIN + kt * BK + lhalf);
        ra[0] = p[0]; ra[1] = p[1]; ra[2] = p[2]; ra[3] = p[3];
    };
    // --- A tile: regs -> bf16 -> LDS (v_perm packing) -----------------------
    auto stageA = [&](int buf, const float4 ra[4]) {
        unsigned pk[8];
#pragma unroll
        for (int v = 0; v < 4; ++v) {
            pk[v * 2 + 0] = pack2(ra[v].x, ra[v].y);
            pk[v * 2 + 1] = pack2(ra[v].z, ra[v].w);
        }
        uint4* ad = (uint4*)&As[buf][lrow * LDT + lhalf];
        ad[0] = make_uint4(pk[0], pk[1], pk[2], pk[3]);
        ad[1] = make_uint4(pk[4], pk[5], pk[6], pk[7]);
    };

    // A fragment (16x32 bf16): lane L -> m=L&15; K-base 0 (L<16) / 8 (L>=16);
    // two contiguous 8-element chunks at kb and kb+16.
    auto fragA = [&](const unsigned short* base, int mi) -> v16bf {
        int row = wm + mi * 16 + (lane & 15);
        int kb  = (lane < 16) ? 0 : 8;
        v8u lo = *(const v8u*)&base[row * LDT + kb];
        v8u hi = *(const v8u*)&base[row * LDT + kb + 16];
        v16u v;
#pragma unroll
        for (int e = 0; e < 8; ++e) { v[e] = lo[e]; v[e + 8] = hi[e]; }
        return __builtin_bit_cast(v16bf, v);
    };
    // B fragment (32x16 bf16): lane L -> n=L&15; K-base 0 (L<16) / 8 (L>=16);
    // two contiguous 8-element chunks at kb and kb+16 of Bs[n][k].
    auto fragB = [&](const unsigned short* base, int ni) -> v16bf {
        int row = wn + ni * 16 + (lane & 15);
        int kb  = (lane < 16) ? 0 : 8;
        v8u lo = *(const v8u*)&base[row * LDT + kb];
        v8u hi = *(const v8u*)&base[row * LDT + kb + 16];
        v16u v;
#pragma unroll
        for (int e = 0; e < 8; ++e) { v[e] = lo[e]; v[e + 8] = hi[e]; }
        return __builtin_bit_cast(v16bf, v);
    };

    const int KT = DIN / BK;                  // 128 k-steps

    // Prologue: fill buffer 0.
    {
        float4 ra[4];
        asyncB(0, 0);
        loadA(0, ra);
        stageA(0, ra);
        waitAsync();
        __syncthreads();
    }

    for (int kt = 0; kt < KT; ++kt) {
        const int cur = kt & 1;
        const int nxt = cur ^ 1;
        float4 ra[4];
        const bool more = (kt + 1 < KT);
        if (more) {
            asyncB(kt + 1, nxt);              // fills Bs[nxt] directly in LDS
            loadA(kt + 1, ra);                // overlaps with compute below
            if (kt + 2 < KT)                  // L2 prefetch hint for A stream
                __builtin_prefetch(X + (size_t)(mBase + lrow) * DIN + (kt + 2) * BK + lhalf, 0, 1);
        }

        const unsigned short* a = As[cur];
        const unsigned short* b = Bs[cur];
        v16bf af0 = fragA(a, 0), af1 = fragA(a, 1);
#pragma unroll
        for (int ni = 0; ni < 4; ++ni) {
            v16bf bfb = fragB(b, ni);
            acc[0][ni] = __builtin_amdgcn_wmma_f32_16x16x32_bf16(
                false, af0, false, bfb, (short)0, acc[0][ni], false, false);
            acc[1][ni] = __builtin_amdgcn_wmma_f32_16x16x32_bf16(
                false, af1, false, bfb, (short)0, acc[1][ni], false, false);
        }

        if (more) {
            stageA(nxt, ra);
            waitAsync();                      // Bs[nxt] fully in LDS
        }
        __syncthreads();                      // single barrier per iteration
    }

    // Epilogue: C/D layout -> VGPR r: lanes 0-15 (M=r, N=lane), lanes 16-31 (M=r+8).
#pragma unroll
    for (int ni = 0; ni < 4; ++ni) {
        int o = nBase + wn + ni * 16 + (lane & 15);
        float cv = cvec[o];
#pragma unroll
        for (int mi = 0; mi < 2; ++mi) {
            int rbase = mBase + wm + mi * 16 + ((lane < 16) ? 0 : 8);
#pragma unroll
            for (int r = 0; r < 8; ++r)
                Out[(size_t)(rbase + r) * DOUT + o] = acc[mi][ni][r] + cv;
        }
    }
}

// ---------------------------------------------------------------------------
extern "C" void kernel_launch(void* const* d_in, const int* in_sizes, int n_in,
                              void* d_out, int out_size, void* d_ws, size_t ws_size,
                              hipStream_t stream) {
    const float* x    = (const float*)d_in[0];
    const float* w1a  = (const float*)d_in[1];
    const float* w1b  = (const float*)d_in[2];
    const float* w2a  = (const float*)d_in[3];
    const float* w2b  = (const float*)d_in[4];
    const float* bw   = (const float*)d_in[5];
    const float* bias = (const float*)d_in[6];
    const float* dora = (const float*)d_in[7];
    float* out = (float*)d_out;

    // workspace layout: w1 | w2 | mag | cvec | (pad to 64KB) | Wt bf16 (32MB)
    char* ws = (char*)d_ws;
    float* w1   = (float*)ws;
    float* w2   = w1 + 4096;
    float* mag  = w2 + 4096;
    float* cvec = mag + 4096;
    unsigned short* wt = (unsigned short*)(ws + 65536);

    k_factors<<<1, 256, 0, stream>>>(w1a, w1b, w2a, w2b, w1, w2);
    k_norm<<<DOUT, 256, 0, stream>>>(bw, bias, dora, w1, w2, mag, cvec);
    k_wt<<<DOUT, 256, 0, stream>>>(bw, w1, w2, mag, wt);
    dim3 grid(DOUT / BN, MTOT / BM);          // 32 x 128
    k_gemm<<<grid, 256, 0, stream>>>(x, wt, cvec, out);
}
